// Policy_3324304687819
// MI455X (gfx1250) — matrix operand order repaired and checked by reading, not compiled, not run on hardware
//
#include <hip/hip_runtime.h>
#include <hip/hip_bf16.h>

// ---------- types ----------
typedef __attribute__((ext_vector_type(16))) __bf16          v16bf;
typedef __attribute__((ext_vector_type(8)))  float           v8f;
typedef __attribute__((ext_vector_type(8)))  unsigned short  u16x8;
typedef __attribute__((ext_vector_type(16))) unsigned short  u16x16;
typedef int v4i __attribute__((vector_size(16)));
typedef __attribute__((address_space(1))) v4i* gptr4;        // global int4*
typedef __attribute__((address_space(3))) v4i* lptr4;        // LDS int4*

#define TSTEPS 512
#define HDIM   2048
#define G4     8192
#define NOUT   4096
#define NBLK   64          // persistent workgroups (64 x 256 thr = 512 waves)
#define XPITCH 264         // 256 + 8 pad (keeps 16B align, staggers LDS banks)

#if defined(__has_builtin)
#if __has_builtin(__builtin_amdgcn_global_load_async_to_lds_b128) && \
    __has_builtin(__builtin_amdgcn_s_wait_asynccnt)
#define HAVE_ASYNC_LDS 1
#endif
#endif
#ifndef HAVE_ASYNC_LDS
#define HAVE_ASYNC_LDS 0
#endif

// ---------- helpers ----------
__device__ __forceinline__ unsigned short f32_to_bf16(float f) {
    unsigned u = __float_as_uint(f);
    u += 0x7FFFu + ((u >> 16) & 1u);     // round-to-nearest-even
    return (unsigned short)(u >> 16);
}

__device__ __forceinline__ v16bf load_frag(const unsigned short* p0,
                                           const unsigned short* p1) {
    u16x8 a = *reinterpret_cast<const u16x8*>(p0);
    u16x8 b = *reinterpret_cast<const u16x8*>(p1);
    u16x16 u = __builtin_shufflevector(a, b, 0,1,2,3,4,5,6,7,8,9,10,11,12,13,14,15);
    return __builtin_bit_cast(v16bf, u);
}

__device__ __forceinline__ v8f wmma_bf16(v16bf a, v16bf b, v8f c) {
    return __builtin_amdgcn_wmma_f32_16x16x32_bf16(false, a, false, b,
                                                   (short)0, c, false, false);
}

__device__ __forceinline__ float sigm(float x) { return 1.f / (1.f + expf(-x)); }

__device__ __forceinline__ void grid_barrier(unsigned* cnt, unsigned* gen, unsigned nblk) {
    __syncthreads();
    __threadfence();
    if (threadIdx.x == 0) {
        unsigned g = __hip_atomic_load(gen, __ATOMIC_RELAXED, __HIP_MEMORY_SCOPE_AGENT);
        unsigned arrived = __hip_atomic_fetch_add(cnt, 1u, __ATOMIC_ACQ_REL,
                                                  __HIP_MEMORY_SCOPE_AGENT) + 1u;
        if (arrived == nblk) {
            __hip_atomic_store(cnt, 0u, __ATOMIC_RELAXED, __HIP_MEMORY_SCOPE_AGENT);
            __hip_atomic_store(gen, g + 1u, __ATOMIC_RELEASE, __HIP_MEMORY_SCOPE_AGENT);
        } else {
            while (__hip_atomic_load(gen, __ATOMIC_ACQUIRE, __HIP_MEMORY_SCOPE_AGENT) == g)
                __builtin_amdgcn_s_sleep(1);
        }
    }
    __syncthreads();
    __threadfence();
}

// ---------- kernels ----------
__global__ void k_cvt_bf16(const float* __restrict__ src,
                           unsigned short* __restrict__ dst, int n) {
    int i = blockIdx.x * blockDim.x + threadIdx.x;
    int stride = gridDim.x * blockDim.x;
    for (; i < n; i += stride) dst[i] = f32_to_bf16(src[i]);
}

__global__ void k_embed(const int* __restrict__ guesses, const int* __restrict__ feedbacks,
                        const float* __restrict__ ge, const float* __restrict__ fe,
                        float* __restrict__ Xf, unsigned short* __restrict__ Xb) {
    int t = blockIdx.x;
    int g = guesses[t], f = feedbacks[t];
    for (int e = threadIdx.x; e < 1024; e += blockDim.x) {
        float a = ge[(size_t)g * 1024 + e];
        float b = fe[(size_t)f * 1024 + e];
        Xf[(size_t)t * HDIM + e]         = a;
        Xf[(size_t)t * HDIM + 1024 + e]  = b;
        Xb[(size_t)t * HDIM + e]         = f32_to_bf16(a);
        Xb[(size_t)t * HDIM + 1024 + e]  = f32_to_bf16(b);
    }
}

__global__ void k_init(unsigned* __restrict__ bar) {
    if (threadIdx.x < 2) bar[threadIdx.x] = 0u;
}

// GX[512,8192] = X @ W_ih^T + b_ih + b_hh   (bf16 WMMA, f32 accumulate)
// 512 blocks x 8 waves. All 8 waves of a block share one M-tile of X, so the
// X chunk (16x256) is staged once per block into LDS (async global->LDS on
// CDNA5), A-fragments then broadcast from LDS; B streams from L2.
__global__ void __launch_bounds__(256) k_gemm_x(
        const unsigned short* __restrict__ Xb, const unsigned short* __restrict__ Wihb,
        const float* __restrict__ bih, const float* __restrict__ bhh,
        float* __restrict__ GX) {
    __shared__ __align__(16) unsigned short xs[16 * XPITCH];   // 8.25 KB
    const int wave = threadIdx.x >> 5;
    const int lane = threadIdx.x & 31;
    const int job  = blockIdx.x * 8 + wave;       // 4096 jobs
    const int mTile  = job >> 7;                  // uniform within a block
    const int nTile0 = (job & 127) << 2;          // [0,512) step 4
    const int n16  = lane & 31 & 15;
    const int half = lane >> 4;
    const int base8  = half * 8;
    const int base16 = half * 16;

    // staging assignment: thread copies 16 elems (32B) of the 16x256 tile
    const int srow  = threadIdx.x >> 4;           // 0..15
    const int scol  = (threadIdx.x & 15) * 16;    // 0..240
    const unsigned short* gbase = Xb + (size_t)(mTile * 16 + srow) * HDIM + scol;
    unsigned short* lbase = &xs[srow * XPITCH + scol];

    v8f acc[4] = { {}, {}, {}, {} };
    for (int kc = 0; kc < HDIM; kc += 256) {
#if HAVE_ASYNC_LDS
        __builtin_amdgcn_global_load_async_to_lds_b128(
            (gptr4)(gbase + kc), (lptr4)lbase, 0, 0);
        __builtin_amdgcn_global_load_async_to_lds_b128(
            (gptr4)(gbase + kc + 8), (lptr4)(lbase + 8), 0, 0);
        __builtin_amdgcn_s_wait_asynccnt(0);
#else
        *reinterpret_cast<u16x8*>(lbase)     = *reinterpret_cast<const u16x8*>(gbase + kc);
        *reinterpret_cast<u16x8*>(lbase + 8) = *reinterpret_cast<const u16x8*>(gbase + kc + 8);
#endif
        __syncthreads();
        #pragma unroll
        for (int k0 = 0; k0 < 256; k0 += 32) {
            v16bf a = load_frag(&xs[n16 * XPITCH + k0 + base8],
                                &xs[n16 * XPITCH + k0 + 16 + base8]);
            #pragma unroll
            for (int j = 0; j < 4; ++j) {
                const unsigned short* brow =
                    Wihb + (size_t)((nTile0 + j) * 16 + n16) * HDIM + kc + k0 + base8;
                __builtin_prefetch(brow + 256, 0, 1);
                v16bf b = load_frag(brow, brow + 16);
                acc[j] = wmma_bf16(a, b, acc[j]);
            }
        }
        __syncthreads();
    }
    const int mOff = half * 8;
    #pragma unroll
    for (int j = 0; j < 4; ++j) {
        int r = (nTile0 + j) * 16 + n16;
        float bias = bih[r] + bhh[r];
        #pragma unroll
        for (int v = 0; v < 8; ++v)
            GX[(size_t)(mTile * 16 + mOff + v) * G4 + r] = acc[j][v] + bias;
    }
}

// Persistent LSTM: 64 blocks x 8 waves = 512 waves, 16 gate rows each.
// h lives in bf16 in EVERY block's LDS; after the single grid barrier per
// step each block redundantly computes the full 2048-wide elementwise update
// from the shared gates array and refreshes its LDS copy -> one barrier/step,
// zero global h traffic, B-fragments are conflict-free ds_load broadcasts.
__global__ void __launch_bounds__(256) k_lstm(
        const unsigned short* __restrict__ Whhb, const float* __restrict__ GX,
        const float* __restrict__ Xf, float* __restrict__ hf,
        float* __restrict__ gates, unsigned* __restrict__ bar) {
    __shared__ __align__(16) unsigned short hlds[HDIM];       // 4 KB
    const int wave = threadIdx.x >> 5;
    const int lane = threadIdx.x & 31;
    const int gwave = blockIdx.x * 8 + wave;      // [0,512)
    const int rowBase = gwave * 16;
    const int n16  = lane & 15;
    const int half = lane >> 4;
    const int base8  = half * 8;
    const int base16 = half * 16;
    const unsigned short* arow = Whhb + (size_t)(rowBase + n16) * HDIM;

    for (int j = threadIdx.x; j < HDIM; j += 256) hlds[j] = 0;  // h0 = 0
    __syncthreads();

    for (int t = 0; t < TSTEPS; ++t) {
        v8f acc0 = {}, acc1 = {};
        #pragma unroll 4
        for (int k0 = 0; k0 < HDIM; k0 += 64) {
            __builtin_prefetch(arow + k0 + 512, 0, 1);
            v16bf a0 = load_frag(arow + k0 + base8,      arow + k0 + 16 + base8);
            v16bf b0 = load_frag(&hlds[k0 + base8],      &hlds[k0 + 16 + base8]);
            acc0 = wmma_bf16(a0, b0, acc0);
            v16bf a1 = load_frag(arow + k0 + 32 + base8, arow + k0 + 48 + base8);
            v16bf b1 = load_frag(&hlds[k0 + 32 + base8], &hlds[k0 + 48 + base8]);
            acc1 = wmma_bf16(a1, b1, acc1);
        }
        if (n16 == 0) {                            // lanes 0 and 16 hold column 0
            #pragma unroll
            for (int v = 0; v < 8; ++v) {
                int r = rowBase + half * 8 + v;
                gates[r] = acc0[v] + acc1[v] + GX[(size_t)t * G4 + r];
            }
        }
        grid_barrier(bar, bar + 1, NBLK);          // gates visible device-wide

        #pragma unroll
        for (int q = 0; q < 8; ++q) {              // redundant per-block update
            int j = threadIdx.x + 256 * q;         // PyTorch gate order i,f,g,o
            float iG = gates[j];
            float fG = gates[HDIM + j];
            float gG = gates[2 * HDIM + j];
            float oG = gates[3 * HDIM + j];
            float cin = (t > 0) ? Xf[(size_t)t * HDIM + j] : 0.f;  // reference quirk
            float c = sigm(fG) * cin + sigm(iG) * tanhf(gG);
            float h = sigm(oG) * tanhf(c);
            hlds[j] = f32_to_bf16(h);
            if (t == TSTEPS - 1 && blockIdx.x == 0) hf[j] = h;
        }
        __syncthreads();                           // hlds ready for next step
    }
}

// logits[4096] = h @ W_fc^T + b_fc : one wave per row, coalesced
__global__ void __launch_bounds__(256) k_fc(
        const float* __restrict__ Wfc, const float* __restrict__ bfc,
        const float* __restrict__ hf, float* __restrict__ logits) {
    const int wave = threadIdx.x >> 5;
    const int lane = threadIdx.x & 31;
    const int row  = blockIdx.x * 8 + wave;       // 512 blocks -> 4096 rows
    const float* wrow = Wfc + (size_t)row * HDIM;
    float s = 0.f;
    for (int k = lane; k < HDIM; k += 32) s += wrow[k] * hf[k];
    #pragma unroll
    for (int off = 16; off; off >>= 1) s += __shfl_xor(s, off, 32);
    if (lane == 0) logits[row] = s + bfc[row];
}

__global__ void k_softmax(const float* __restrict__ logits, float* __restrict__ out) {
    __shared__ float sm[32];
    const int tid = threadIdx.x, lane = tid & 31, wv = tid >> 5;
    float m = -3.4e38f;
    for (int i = tid; i < NOUT; i += 1024) m = fmaxf(m, logits[i]);
    #pragma unroll
    for (int off = 16; off; off >>= 1) m = fmaxf(m, __shfl_xor(m, off, 32));
    if (lane == 0) sm[wv] = m;
    __syncthreads();
    if (wv == 0) {
        float v = sm[lane];
        #pragma unroll
        for (int off = 16; off; off >>= 1) v = fmaxf(v, __shfl_xor(v, off, 32));
        if (lane == 0) sm[0] = v;
    }
    __syncthreads();
    m = sm[0];
    __syncthreads();
    float s = 0.f;
    for (int i = tid; i < NOUT; i += 1024) s += expf(logits[i] - m);
    #pragma unroll
    for (int off = 16; off; off >>= 1) s += __shfl_xor(s, off, 32);
    if (lane == 0) sm[wv] = s;
    __syncthreads();
    if (wv == 0) {
        float v = sm[lane];
        #pragma unroll
        for (int off = 16; off; off >>= 1) v += __shfl_xor(v, off, 32);
        if (lane == 0) sm[0] = v;
    }
    __syncthreads();
    float inv = 1.f / sm[0];
    for (int i = tid; i < NOUT; i += 1024) out[i] = expf(logits[i] - m) * inv;
}

// ---------- launch ----------
extern "C" void kernel_launch(void* const* d_in, const int* in_sizes, int n_in,
                              void* d_out, int out_size, void* d_ws, size_t ws_size,
                              hipStream_t stream) {
    (void)in_sizes; (void)n_in; (void)out_size; (void)ws_size;
    const int*   guesses   = (const int*)  d_in[0];
    const int*   feedbacks = (const int*)  d_in[1];
    const float* ge        = (const float*)d_in[2];
    const float* fe        = (const float*)d_in[3];
    const float* Wih       = (const float*)d_in[4];
    const float* Whh       = (const float*)d_in[5];
    const float* bih       = (const float*)d_in[6];
    const float* bhh       = (const float*)d_in[7];
    const float* Wfc       = (const float*)d_in[8];
    const float* bfc       = (const float*)d_in[9];

    char* ws = (char*)d_ws;
    size_t o = 0;
    auto carve = [&](size_t bytes) { char* p = ws + o; o = (o + bytes + 255) & ~(size_t)255; return p; };
    unsigned short* Xb   = (unsigned short*)carve((size_t)TSTEPS * HDIM * 2);
    float*          Xf   = (float*)         carve((size_t)TSTEPS * HDIM * 4);
    unsigned short* Wihb = (unsigned short*)carve((size_t)G4 * HDIM * 2);
    unsigned short* Whhb = (unsigned short*)carve((size_t)G4 * HDIM * 2);
    float*          GX   = (float*)         carve((size_t)TSTEPS * G4 * 4);
    float*          hf   = (float*)         carve((size_t)HDIM * 4);
    float*          gate = (float*)         carve((size_t)G4 * 4);
    float*          lg   = (float*)         carve((size_t)NOUT * 4);
    unsigned*       bar  = (unsigned*)      carve(256);

    const int NW = G4 * HDIM;   // 16,777,216
    k_cvt_bf16<<<2048, 256, 0, stream>>>(Wih, Wihb, NW);
    k_cvt_bf16<<<2048, 256, 0, stream>>>(Whh, Whhb, NW);
    k_embed   <<<TSTEPS, 256, 0, stream>>>(guesses, feedbacks, ge, fe, Xf, Xb);
    k_init    <<<1, 64, 0, stream>>>(bar);
    k_gemm_x  <<<512, 256, 0, stream>>>(Xb, Wihb, bih, bhh, GX);
    k_lstm    <<<NBLK, 256, 0, stream>>>(Whhb, GX, Xf, hf, gate, bar);
    k_fc      <<<512, 256, 0, stream>>>(Wfc, bfc, hf, lg);
    k_softmax <<<1, 1024, 0, stream>>>(lg, (float*)d_out);
}
